// EGMFA_12249246728581
// MI455X (gfx1250) — matrix operand order repaired and checked by reading, not compiled, not run on hardware
//
#include <hip/hip_runtime.h>
#include <hip/hip_bf16.h>

// ---------------------------------------------------------------------------
// EGMFA on MI455X (gfx1250):
//   Key identity: sum_{h,w} irfft2(rfft2(p)*rfft2(f)) = (sum p) * (sum f)
//   => asm[b,c] = S_pred[b] * S_feat[b,c]; no FFT needed.
//   Kernel 1: pred_reduce   (8 blocks)   bilinear-upsample+sigmoid on the fly,
//             analytic 3x3-conv sum via boundary accumulators.
//   Kernel 2: feat_reduce   (512 blocks) single-pass 134MB stream (roofline
//             ~5.8us @ 23.3 TB/s), analytic conv sums (3x3 / 1x11 / 11x1).
//   Kernel 3: tail_kernel   (1 block, 8 wave32 waves) all small GEMMs via
//             v_wmma_f32_16x16x32_f16, LN/sigmoid/gating fused in LDS.
// ---------------------------------------------------------------------------

typedef __attribute__((ext_vector_type(16))) _Float16 v16h;
typedef __attribute__((ext_vector_type(8)))  float    v8f;

__device__ __forceinline__ float sigmoidf(float x) { return 1.0f / (1.0f + __expf(-x)); }

// ---------------------------------------------------------------------------
// Kernel 1: pred branch reduction. One block per batch image.
// pred raw: (b,1,128,128) -> bilinear x2 -> sigmoid s -> a = s*(1-sig(s))+s
// S_pred[b] = sum(s) + sum(dwconv3x3(a, pad1)) + 65536*dw_b
// sum(conv) computed analytically from {S_a, R0, R255, C0, C255, corners}.
// ---------------------------------------------------------------------------
__global__ void __launch_bounds__(256) pred_reduce(
    const float* __restrict__ pred, const float* __restrict__ dw_w,
    const float* __restrict__ dw_b, float* __restrict__ spred) {
  const int b = blockIdx.x;
  const int tid = threadIdx.x;
  const float* p = pred + (size_t)b * 16384u;

  float acc[10];
#pragma unroll
  for (int i = 0; i < 10; ++i) acc[i] = 0.0f;

  for (int i = tid; i < 65536; i += 256) {
    const int oh = i >> 8, ow = i & 255;
    // jax.image.resize bilinear: src = (o+0.5)/2 - 0.5, edge-clamped taps
    const float sh = (oh + 0.5f) * 0.5f - 0.5f;
    const float sw = (ow + 0.5f) * 0.5f - 0.5f;
    int h0 = (int)floorf(sh); const float fh = sh - (float)h0;
    int w0 = (int)floorf(sw); const float fw = sw - (float)w0;
    const int h0c = min(max(h0, 0), 127), h1c = min(max(h0 + 1, 0), 127);
    const int w0c = min(max(w0, 0), 127), w1c = min(max(w0 + 1, 0), 127);
    const float v00 = p[h0c * 128 + w0c], v01 = p[h0c * 128 + w1c];
    const float v10 = p[h1c * 128 + w0c], v11 = p[h1c * 128 + w1c];
    const float u = (1.0f - fh) * ((1.0f - fw) * v00 + fw * v01) +
                    fh * ((1.0f - fw) * v10 + fw * v11);
    const float s = sigmoidf(u);
    const float a = s * (1.0f - sigmoidf(s)) + s;
    acc[0] += s;                        // sum of sigmoid(pred)
    acc[1] += a;                        // S_a
    if (oh == 0)   acc[2] += a;         // R0
    if (oh == 255) acc[3] += a;         // R255
    if (ow == 0)   acc[4] += a;         // C0
    if (ow == 255) acc[5] += a;         // C255
    if (oh == 0   && ow == 0)   acc[6] += a;   // a[0][0]
    if (oh == 0   && ow == 255) acc[7] += a;   // a[0][255]
    if (oh == 255 && ow == 0)   acc[8] += a;   // a[255][0]
    if (oh == 255 && ow == 255) acc[9] += a;   // a[255][255]
  }

  __shared__ float red[256];
  __shared__ float tot[10];
  for (int k = 0; k < 10; ++k) {
    red[tid] = acc[k];
    __syncthreads();
    for (int s = 128; s > 0; s >>= 1) {
      if (tid < s) red[tid] += red[tid + s];
      __syncthreads();
    }
    if (tid == 0) tot[k] = red[0];
    __syncthreads();
  }

  if (tid == 0) {
    float out = tot[0];
    for (int kh = 0; kh < 3; ++kh)
      for (int kw = 0; kw < 3; ++kw) {
        float rect = tot[1];
        if (kh == 0) rect -= tot[3]; else if (kh == 2) rect -= tot[2];
        if (kw == 0) rect -= tot[5]; else if (kw == 2) rect -= tot[4];
        if (kh == 0 && kw == 0) rect += tot[9];
        if (kh == 0 && kw == 2) rect += tot[8];
        if (kh == 2 && kw == 0) rect += tot[7];
        if (kh == 2 && kw == 2) rect += tot[6];
        out += dw_w[kh * 3 + kw] * rect;
      }
    out += 65536.0f * dw_b[0];
    spred[b] = out;
  }
}

// ---------------------------------------------------------------------------
// Kernel 2: feat branch reduction. One block per (b,c) channel; 134MB read
// exactly once as b128 loads. Channel classes:
//   c<40: identity;  40..47: 3x3 pad1;  48..55: 1x11 padW5;  56..63: 11x1 padH5
// Conv output sums computed analytically (zero-pad boundary corrections).
// ---------------------------------------------------------------------------
__global__ void __launch_bounds__(256) feat_reduce(
    const float* __restrict__ feat,
    const float* __restrict__ inc_hw_w, const float* __restrict__ inc_hw_b,
    const float* __restrict__ inc_w_w,  const float* __restrict__ inc_w_b,
    const float* __restrict__ inc_h_w,  const float* __restrict__ inc_h_b,
    float* __restrict__ sfeat) {
  const int blk = blockIdx.x;          // b*64 + c
  const int c = blk & 63;
  const int tid = threadIdx.x;
  const int cls = (c < 40) ? 0 : (c < 48) ? 1 : (c < 56) ? 2 : 3;
  const float4* x4 = (const float4*)(feat + (size_t)blk * 65536u);

  float acc[11];
#pragma unroll
  for (int i = 0; i < 11; ++i) acc[i] = 0.0f;

  for (int i = tid; i < 16384; i += 256) {
    const float4 v = x4[i];
    const int h = i >> 6;
    const int wb = (i & 63) << 2;      // w of component .x (group of 4 cols)
    const float rowsum = v.x + v.y + v.z + v.w;
    acc[0] += rowsum;                  // S
    if (cls == 1) {
      if (h == 0)   acc[1] += rowsum;  // R0
      if (h == 255) acc[2] += rowsum;  // R255
      if (wb == 0) {                   // col 0
        acc[3] += v.x;
        if (h == 0)   acc[5] += v.x;   // a[0][0]
        if (h == 255) acc[7] += v.x;   // a[255][0]
      }
      if (wb == 252) {                 // col 255
        acc[4] += v.w;
        if (h == 0)   acc[6] += v.w;   // a[0][255]
        if (h == 255) acc[8] += v.w;   // a[255][255]
      }
    } else if (cls == 2) {             // need cols 0..4 and 251..255
      if (wb == 0)        { acc[1] += v.x; acc[2] += v.y; acc[3] += v.z; acc[4] += v.w; }
      else if (wb == 4)   { acc[5] += v.x; }
      else if (wb == 248) { acc[6] += v.w; }                       // col 251
      else if (wb == 252) { acc[7] += v.x; acc[8] += v.y; acc[9] += v.z; acc[10] += v.w; }
    } else if (cls == 3) {             // need rows 0..4 and 251..255
      if (h < 5)        acc[1 + h] += rowsum;
      else if (h >= 251) acc[6 + (h - 251)] += rowsum;
    }
  }

  __shared__ float red[256];
  __shared__ float tot[11];
  for (int k = 0; k < 11; ++k) {
    red[tid] = acc[k];
    __syncthreads();
    for (int s = 128; s > 0; s >>= 1) {
      if (tid < s) red[tid] += red[tid + s];
      __syncthreads();
    }
    if (tid == 0) tot[k] = red[0];
    __syncthreads();
  }

  if (tid == 0) {
    float r;
    if (cls == 0) {
      r = tot[0];
    } else if (cls == 1) {
      const int g = c - 40;
      const float* w3 = inc_hw_w + g * 9;
      float s = 0.0f;
      for (int kh = 0; kh < 3; ++kh)
        for (int kw = 0; kw < 3; ++kw) {
          float rect = tot[0];
          if (kh == 0) rect -= tot[2]; else if (kh == 2) rect -= tot[1];
          if (kw == 0) rect -= tot[4]; else if (kw == 2) rect -= tot[3];
          if (kh == 0 && kw == 0) rect += tot[8];
          if (kh == 0 && kw == 2) rect += tot[7];
          if (kh == 2 && kw == 0) rect += tot[6];
          if (kh == 2 && kw == 2) rect += tot[5];
          s += w3[kh * 3 + kw] * rect;
        }
      r = s + 65536.0f * inc_hw_b[g];
    } else {
      const int g = (cls == 2) ? (c - 48) : (c - 56);
      const float* w11 = (cls == 2) ? (inc_w_w + g * 11) : (inc_h_w + g * 11);
      const float bias = (cls == 2) ? inc_w_b[g] : inc_h_b[g];
      float s = 0.0f;
      for (int k = 0; k < 11; ++k) {
        float rect = tot[0];
        if (k < 5) {                              // exclude trailing cols/rows k+251..255
          for (int j = k; j < 5; ++j) rect -= tot[6 + j];
        } else if (k > 5) {                       // exclude leading cols/rows 0..k-6
          for (int j = 0; j <= k - 6; ++j) rect -= tot[1 + j];
        }
        s += w11[k] * rect;
      }
      r = s + 65536.0f * bias;
    }
    sfeat[blk] = r;
  }
}

// ---------------------------------------------------------------------------
// WMMA fragment builders (v_wmma_f32_16x16x32_f16, wave32).
// A fragment (16x32 f16), contiguous k order (same as B):
//   lane<16 (M=lane):   elems i -> K=ks*32+i (0..15)
//   lane>=16 (M=lane-16): elems i -> K=ks*32+16+i (16..31)
// B fragment (32x16 f16), B[k][n] = W[n*K + k] (y = x @ W^T, W row-major NxK):
//   lane<16: column N=lane, elems i -> K=ks*32+i ; lane>=16: K=ks*32+16+i
// ---------------------------------------------------------------------------
__device__ __forceinline__ v16h load_A_frag(const float* A /*[16][64] LDS*/, int lane, int ks) {
  const int m = lane & 15;
  const int kb = ks * 32 + ((lane & 16) ? 16 : 0);
  v16h a;
#pragma unroll
  for (int i = 0; i < 8; ++i) a[i] = (_Float16)A[m * 64 + kb + i];
#pragma unroll
  for (int i = 0; i < 8; ++i) a[8 + i] = (_Float16)A[m * 64 + kb + 8 + i];
  return a;
}

__device__ __forceinline__ v16h load_B_frag(const float* __restrict__ W, int lane, int ks,
                                            int ntile, int K) {
  const int n = ntile * 16 + (lane & 15);
  const int kb = ks * 32 + ((lane & 16) ? 16 : 0);
  v16h b;
#pragma unroll
  for (int i = 0; i < 16; ++i) b[i] = (_Float16)W[n * K + kb + i];
  return b;
}

struct TailParams {
  const float *pt_w, *pt_b, *ht_w, *ht_b;
  const float *hg_w, *hg_b, *pg_w, *pg_b;
  const float *fc_w, *fc_b;
  const float *ln_pin_g, *ln_pin_b, *ln_hin_g, *ln_hin_b;
  const float *ln_pout_g, *ln_pout_b, *ln_hout_g, *ln_hout_b;
  const float *ln_fc_g, *ln_fc_b;
};

// ---------------------------------------------------------------------------
// Kernel 3: fused tail. One block, 256 threads = 8 wave32 waves.
// All GEMMs are M=8 (padded to 16) x K=64 x N in {128,64}; each wave owns one
// 16-wide N tile and runs 2 K-steps of v_wmma_f32_16x16x32_f16.
// ---------------------------------------------------------------------------
__global__ void __launch_bounds__(256) tail_kernel(
    const float* __restrict__ spred, const float* __restrict__ sfeat,
    const float* __restrict__ head, TailParams P, float* __restrict__ out) {
  __shared__ float af[16][64];     // asm features (padded)
  __shared__ float ha[16][64];     // head features (padded)
  __shared__ float pf[8][128];
  __shared__ float hf[8][128];
  __shared__ float gate[16][64];
  __shared__ float hgraw[8][64];
  __shared__ float pgraw[8][64];
  __shared__ float upd[16][64];
  __shared__ float updraw[8][64];
  __shared__ float mu[32], rstd[32];

  const int tid = threadIdx.x;
  const int lane = tid & 31;
  const int wv = tid >> 5;

  // asm[b,c] = S_pred[b] * S_feat[b,c]  (the collapsed FFT block)
  for (int i = tid; i < 16 * 64; i += 256) {
    const int m = i >> 6, k = i & 63;
    af[m][k] = (m < 8) ? spred[m] * sfeat[m * 64 + k] : 0.0f;
    ha[m][k] = (m < 8) ? head[m * 64 + k] : 0.0f;
  }
  __syncthreads();

  // pf = af @ pt_w^T + pt_b ; hf = ha @ ht_w^T + ht_b   (N=128: 8 tiles = 8 waves)
  {
    v8f dp = {}; v8f dh = {};
#pragma unroll
    for (int ks = 0; ks < 2; ++ks) {
      const v16h aA = load_A_frag(&af[0][0], lane, ks);
      const v16h aH = load_A_frag(&ha[0][0], lane, ks);
      const v16h bP = load_B_frag(P.pt_w, lane, ks, wv, 64);
      const v16h bH = load_B_frag(P.ht_w, lane, ks, wv, 64);
      dp = __builtin_amdgcn_wmma_f32_16x16x32_f16(false, aA, false, bP, (short)0, dp, false, false);
      dh = __builtin_amdgcn_wmma_f32_16x16x32_f16(false, aH, false, bH, (short)0, dh, false, false);
    }
    if (lane < 16) {  // lanes 0-15 hold D rows 0..7 (VGPR j = row j)
      const int n = wv * 16 + lane;
#pragma unroll
      for (int j = 0; j < 8; ++j) {
        pf[j][n] = dp[j] + P.pt_b[n];
        hf[j][n] = dh[j] + P.ht_b[n];
      }
    }
  }
  __syncthreads();

  // gate = hf_in * pf_in
  for (int i = tid; i < 16 * 64; i += 256) {
    const int m = i >> 6, k = i & 63;
    gate[m][k] = (m < 8) ? hf[m][k] * pf[m][k] : 0.0f;
  }
  __syncthreads();

  // hgraw = gate @ hg_w^T + hg_b (waves 0-3) ; pgraw = gate @ pg_w^T + pg_b (waves 4-7)
  {
    const int nt = wv & 3;
    const float* W = (wv < 4) ? P.hg_w : P.pg_w;
    const float* B = (wv < 4) ? P.hg_b : P.pg_b;
    float* dst = (wv < 4) ? &hgraw[0][0] : &pgraw[0][0];
    v8f d = {};
#pragma unroll
    for (int ks = 0; ks < 2; ++ks) {
      const v16h a = load_A_frag(&gate[0][0], lane, ks);
      const v16h b = load_B_frag(W, lane, ks, nt, 64);
      d = __builtin_amdgcn_wmma_f32_16x16x32_f16(false, a, false, b, (short)0, d, false, false);
    }
    if (lane < 16) {
      const int n = nt * 16 + lane;
#pragma unroll
      for (int j = 0; j < 8; ++j) dst[j * 64 + n] = d[j] + B[n];
    }
  }
  __syncthreads();

  // LayerNorm stats: tid 0-7 hgraw rows, 8-15 pgraw rows, 16-23 hf[:,64:], 24-31 pf[:,64:]
  if (tid < 32) {
    const int r = tid & 7;
    const float* src = (tid < 8)  ? &hgraw[r][0]
                     : (tid < 16) ? &pgraw[r][0]
                     : (tid < 24) ? &hf[r][64]
                                  : &pf[r][64];
    float m = 0.0f;
    for (int k = 0; k < 64; ++k) m += src[k];
    m *= (1.0f / 64.0f);
    float v = 0.0f;
    for (int k = 0; k < 64; ++k) { const float d = src[k] - m; v += d * d; }
    v *= (1.0f / 64.0f);
    mu[tid] = m;
    rstd[tid] = rsqrtf(v + 1e-5f);
  }
  __syncthreads();

  // upd = sigmoid(LN(pgraw)) * LN(pf_out) + sigmoid(LN(hgraw)) * LN(hf_out)
  for (int i = tid; i < 16 * 64; i += 256) {
    const int m = i >> 6, k = i & 63;
    if (m < 8) {
      const float hgv = sigmoidf((hgraw[m][k] - mu[m])      * rstd[m]      * P.ln_hin_g[k]  + P.ln_hin_b[k]);
      const float pgv = sigmoidf((pgraw[m][k] - mu[8 + m])  * rstd[8 + m]  * P.ln_pin_g[k]  + P.ln_pin_b[k]);
      const float hov =          (hf[m][64+k] - mu[16 + m]) * rstd[16 + m] * P.ln_hout_g[k] + P.ln_hout_b[k];
      const float pov =          (pf[m][64+k] - mu[24 + m]) * rstd[24 + m] * P.ln_pout_g[k] + P.ln_pout_b[k];
      upd[m][k] = pgv * pov + hgv * hov;
    } else {
      upd[m][k] = 0.0f;
    }
  }
  __syncthreads();

  // updraw = upd @ fc_w^T + fc_b (waves 0-3)
  if (wv < 4) {
    v8f d = {};
#pragma unroll
    for (int ks = 0; ks < 2; ++ks) {
      const v16h a = load_A_frag(&upd[0][0], lane, ks);
      const v16h b = load_B_frag(P.fc_w, lane, ks, wv, 64);
      d = __builtin_amdgcn_wmma_f32_16x16x32_f16(false, a, false, b, (short)0, d, false, false);
    }
    if (lane < 16) {
      const int n = wv * 16 + lane;
#pragma unroll
      for (int j = 0; j < 8; ++j) updraw[j][n] = d[j] + P.fc_b[n];
    }
  }
  __syncthreads();

  if (tid < 8) {
    float m = 0.0f;
    for (int k = 0; k < 64; ++k) m += updraw[tid][k];
    m *= (1.0f / 64.0f);
    float v = 0.0f;
    for (int k = 0; k < 64; ++k) { const float d = updraw[tid][k] - m; v += d * d; }
    v *= (1.0f / 64.0f);
    mu[tid] = m;
    rstd[tid] = rsqrtf(v + 1e-5f);
  }
  __syncthreads();

  // out[b,0,f,0,0] = relu(LN(updraw))  -> flat index b*64 + f
  for (int i = tid; i < 512; i += 256) {
    const int m = i >> 6, k = i & 63;
    const float val = (updraw[m][k] - mu[m]) * rstd[m] * P.ln_fc_g[k] + P.ln_fc_b[k];
    out[m * 64 + k] = fmaxf(val, 0.0f);
  }
}

// ---------------------------------------------------------------------------
// Launch. Input order = setup_inputs dict order (params flattened in order).
// ---------------------------------------------------------------------------
extern "C" void kernel_launch(void* const* d_in, const int* in_sizes, int n_in,
                              void* d_out, int out_size, void* d_ws, size_t ws_size,
                              hipStream_t stream) {
  const float* feat     = (const float*)d_in[0];
  const float* head     = (const float*)d_in[1];
  const float* pred     = (const float*)d_in[2];
  const float* dw_w     = (const float*)d_in[3];
  const float* dw_b     = (const float*)d_in[4];
  const float* inc_hw_w = (const float*)d_in[5];
  const float* inc_hw_b = (const float*)d_in[6];
  const float* inc_w_w  = (const float*)d_in[7];
  const float* inc_w_b  = (const float*)d_in[8];
  const float* inc_h_w  = (const float*)d_in[9];
  const float* inc_h_b  = (const float*)d_in[10];

  TailParams P;
  P.pt_w = (const float*)d_in[11];  P.pt_b = (const float*)d_in[12];
  P.ht_w = (const float*)d_in[13];  P.ht_b = (const float*)d_in[14];
  P.pg_w = (const float*)d_in[15];  P.pg_b = (const float*)d_in[16];
  P.hg_w = (const float*)d_in[17];  P.hg_b = (const float*)d_in[18];
  P.fc_w = (const float*)d_in[19];  P.fc_b = (const float*)d_in[20];
  P.ln_pin_g  = (const float*)d_in[21]; P.ln_pin_b  = (const float*)d_in[22];
  P.ln_hin_g  = (const float*)d_in[23]; P.ln_hin_b  = (const float*)d_in[24];
  P.ln_pout_g = (const float*)d_in[25]; P.ln_pout_b = (const float*)d_in[26];
  P.ln_hout_g = (const float*)d_in[27]; P.ln_hout_b = (const float*)d_in[28];
  P.ln_fc_g   = (const float*)d_in[29]; P.ln_fc_b   = (const float*)d_in[30];

  float* spred = (float*)d_ws;       // 8 floats
  float* sfeat = spred + 8;          // 512 floats

  pred_reduce<<<8, 256, 0, stream>>>(pred, dw_w, dw_b, spred);
  feat_reduce<<<512, 256, 0, stream>>>(feat, inc_hw_w, inc_hw_b,
                                       inc_w_w, inc_w_b, inc_h_w, inc_h_b, sfeat);
  tail_kernel<<<1, 256, 0, stream>>>(spred, sfeat, head, P, (float*)d_out);
}
